// MultiHeadAttentionSearch_13778255086129
// MI455X (gfx1250) — matrix operand order repaired and checked by reading, not compiled, not run on hardware
//
#include <hip/hip_runtime.h>

typedef __attribute__((ext_vector_type(16))) _Float16 v16h;
typedef __attribute__((ext_vector_type(8)))  float    v8f;

#define BATCH 4
#define SEQ   1024
#define FEAT  1024
#define NH    16
#define HD    64
#define NOUT  (NH * HD)   // 1024

// ---- fragment load helpers -------------------------------------------------
// A fragment 16x32 f16, contiguous layout (lanes 0-15: K 0-15; +16: 16-31)
__device__ __forceinline__ v16h load_afrag(const _Float16* __restrict__ xp,
                                           int kb) {
  v16h a;
#pragma unroll
  for (int j = 0; j < 4; ++j) {
    a[2 * j]         = xp[kb + 2 * j];
    a[2 * j + 1]     = xp[kb + 2 * j + 1];
    a[8 + 2 * j]     = xp[8 + kb + 2 * j];
    a[8 + 2 * j + 1] = xp[8 + kb + 2 * j + 1];
  }
  return a;
}
// B fragment: 16 contiguous halfs per lane (32B -> b128 pair)
__device__ __forceinline__ v16h load_bfrag(const _Float16* __restrict__ wp) {
  v16h b;
#pragma unroll
  for (int p = 0; p < 16; ++p) b[p] = wp[p];
  return b;
}

// ---------------------------------------------------------------------------
// One-shot f32 -> f16 convert (X staging). 8 elements / thread.
// ---------------------------------------------------------------------------
__global__ __launch_bounds__(256) void cvt_f32_f16_kernel(
    const float* __restrict__ in, _Float16* __restrict__ out, int n) {
  const int idx = (blockIdx.x * 256 + threadIdx.x) * 8;
  if (idx >= n) return;
#pragma unroll
  for (int i = 0; i < 8; ++i) out[idx + i] = (_Float16)in[idx + i];
}

// ---------------------------------------------------------------------------
// W [k][c] f32 -> Wt [c][k] f16 (N-major) so B-fragments are contiguous.
// ---------------------------------------------------------------------------
__global__ __launch_bounds__(256) void cvt_w_transpose_kernel(
    const float* __restrict__ W, _Float16* __restrict__ Wt) {
  const int idx = blockIdx.x * 256 + threadIdx.x;   // 0 .. 1M-1
  const int k = idx >> 10;
  const int c = idx & 1023;
  Wt[(size_t)c * FEAT + k] = (_Float16)W[idx];
}

// ---------------------------------------------------------------------------
// Projection: Y = relu(Xh @ W + b), all-f16 operands, f32 WMMA accumulate.
// One wave computes a 16x64 tile. Software-pipelined: ping-pong fragment
// buffers so next K-step's b128 loads overlap the current 4 WMMAs.
// transposeOut=0: write [B,H,S,D]  (Q, K heads)
// transposeOut=1: write [B,H,D,S]  (V heads, d-major for contiguous PV frags)
// ---------------------------------------------------------------------------
__global__ __launch_bounds__(128) void proj_relu_f16_kernel(
    const _Float16* __restrict__ Xh, const _Float16* __restrict__ Wt,
    const float* __restrict__ bias, _Float16* __restrict__ out,
    int transposeOut) {
  const int lane = threadIdx.x & 31;
  const int tile = blockIdx.x * 4 + (threadIdx.x >> 5);
  const int colTiles = NOUT / 64;           // 16
  const int rt = tile / colTiles;           // 0..255
  const int ct = tile % colTiles;           // 0..15
  const int r0 = rt * 16;
  const int c0 = ct * 64;
  const int hi  = lane >> 4;
  const int kb  = hi * 16;
  const int n   = lane & 15;

  const _Float16* xrow = Xh + (size_t)(r0 + (lane & 15)) * FEAT;
  const _Float16* wrow[4];
#pragma unroll
  for (int t = 0; t < 4; ++t)
    wrow[t] = Wt + (size_t)(c0 + t * 16 + n) * FEAT + 16 * hi;

  v8f acc[4] = {v8f{}, v8f{}, v8f{}, v8f{}};

  // prologue: fragments for K-step 0
  v16h a0 = load_afrag(xrow, kb), a1;
  v16h b0[4], b1[4];
#pragma unroll
  for (int t = 0; t < 4; ++t) b0[t] = load_bfrag(wrow[t]);

  for (int k0 = 0; k0 < FEAT; k0 += 64) {
    const int kn1 = k0 + 32;                // always < FEAT
    const int kn2 = k0 + 64;
    // issue loads for step k0+32 before consuming step k0
    a1 = load_afrag(xrow + kn1, kb);
#pragma unroll
    for (int t = 0; t < 4; ++t) b1[t] = load_bfrag(wrow[t] + kn1);
#pragma unroll
    for (int t = 0; t < 4; ++t)
      acc[t] = __builtin_amdgcn_wmma_f32_16x16x32_f16(
          false, a0, false, b0[t], (short)0, acc[t], false, false);
    // issue loads for step k0+64 before consuming step k0+32
    if (kn2 < FEAT) {
      a0 = load_afrag(xrow + kn2, kb);
#pragma unroll
      for (int t = 0; t < 4; ++t) b0[t] = load_bfrag(wrow[t] + kn2);
    }
#pragma unroll
    for (int t = 0; t < 4; ++t)
      acc[t] = __builtin_amdgcn_wmma_f32_16x16x32_f16(
          false, a1, false, b1[t], (short)0, acc[t], false, false);
  }

  // Epilogue: bias + relu, scatter f16 heads
#pragma unroll
  for (int t = 0; t < 4; ++t) {
    const int c  = c0 + t * 16 + n;
    const float bb = bias[c];
    const int hh = c >> 6;
    const int dd = c & 63;
#pragma unroll
    for (int i = 0; i < 8; ++i) {
      const int r  = r0 + i + 8 * hi;       // C/D layout: M = i + 8*(lane>=16)
      const int bi = r >> 10;
      const int ss = r & 1023;
      float val = acc[t][i] + bb;
      val = val > 0.f ? val : 0.f;
      const size_t headBase = ((size_t)bi * NH + hh);
      if (!transposeOut)
        out[(headBase * SEQ + ss) * HD + dd] = (_Float16)val;
      else
        out[(headBase * HD + dd) * SEQ + ss] = (_Float16)val;
    }
  }
}

// ---------------------------------------------------------------------------
// Flash-style causal attention. One wave (= one block) per (b, h, 16-query
// tile). Online softmax; scores never hit memory. LDS only for the 16x32 P
// re-layout (C-layout -> A-layout) between the QK^T and PV WMMAs.
// V arrives d-major ([B,H,D,S]); its fragments are loaded at the TOP of the
// loop body so the softmax section hides their latency.
// ---------------------------------------------------------------------------
__global__ __launch_bounds__(32) void flash_attn_kernel(
    const _Float16* __restrict__ qh, const _Float16* __restrict__ kh,
    const _Float16* __restrict__ vhT, float* __restrict__ out) {
  __shared__ _Float16 pT[16 * 32];

  const int lane = threadIdx.x & 31;
  const int QT = SEQ / 16;                  // 64
  int tid = blockIdx.x;
  const int qt = tid % QT;  tid /= QT;
  const int h  = tid % NH;  tid /= NH;
  const int b  = tid;

  const _Float16* Q  = qh  + ((size_t)(b * NH + h) * SEQ) * HD;
  const _Float16* K  = kh  + ((size_t)(b * NH + h) * SEQ) * HD;
  const _Float16* Vt = vhT + ((size_t)(b * NH + h) * HD) * SEQ;

  const int row = lane & 15;
  const int hi  = lane >> 4;
  const int kb  = hi * 16;
  const int n   = lane & 15;
  const int q0  = qt * 16;

  // Q A-fragments for D=64 (two 16x32 frags), loaded once.
  const _Float16* qrow = Q + (size_t)(q0 + row) * HD;
  const v16h aq0 = load_afrag(qrow, kb);
  const v16h aq1 = load_afrag(qrow + 32, kb);

  float mrow[8], lrow[8];
  v8f o[4] = {v8f{}, v8f{}, v8f{}, v8f{}};
#pragma unroll
  for (int i = 0; i < 8; ++i) { mrow[i] = -3.0e38f; lrow[i] = 0.f; }

  const int jmax = (q0 + 15) >> 5;          // causal: only needed key blocks
  for (int j = 0; j <= jmax; ++j) {
    const int k0 = j * 32;

    if (j < jmax) {                         // gfx1250 global_prefetch_b8
      __builtin_prefetch(K + (size_t)(k0 + 32 + n + 16 * hi) * HD, 0, 1);
      __builtin_prefetch(Vt + (size_t)(n + 16 * hi) * SEQ + k0 + 32, 0, 1);
    }

    // ---- issue ALL fragment loads up front: V frags retire during softmax --
    v16h bv[4];
#pragma unroll
    for (int t = 0; t < 4; ++t)
      bv[t] = load_bfrag(Vt + (size_t)(t * 16 + n) * SEQ + k0 + 16 * hi);

    v16h bk[4];                             // [g][dchunk]: g*2 + c
#pragma unroll
    for (int g = 0; g < 2; ++g) {
      const _Float16* krow = K + (size_t)(k0 + 16 * g + n) * HD + 16 * hi;
      bk[g * 2]     = load_bfrag(krow);
      bk[g * 2 + 1] = load_bfrag(krow + 32);
    }

    // ---- S = Q K^T for 32 keys: two 16x16 score tiles, K-dim = D = 64 ----
    v8f s0 = {}, s1 = {};
    s0 = __builtin_amdgcn_wmma_f32_16x16x32_f16(false, aq0, false, bk[0],
                                                (short)0, s0, false, false);
    s0 = __builtin_amdgcn_wmma_f32_16x16x32_f16(false, aq1, false, bk[1],
                                                (short)0, s0, false, false);
    s1 = __builtin_amdgcn_wmma_f32_16x16x32_f16(false, aq0, false, bk[2],
                                                (short)0, s1, false, false);
    s1 = __builtin_amdgcn_wmma_f32_16x16x32_f16(false, aq1, false, bk[3],
                                                (short)0, s1, false, false);

    // ---- scale + causal mask (mask input == tril, realized by index) ----
#pragma unroll
    for (int i = 0; i < 8; ++i) {
      const int qg = q0 + i + 8 * hi;
      float v0 = s0[i] * 0.125f;            // 1/sqrt(64)
      float v1 = s1[i] * 0.125f;
      if (k0 + n > qg)      v0 = -1.0e30f;
      if (k0 + 16 + n > qg) v1 = -1.0e30f;
      s0[i] = v0; s1[i] = v1;
    }

    // ---- online softmax: row stats via 16-lane xor-shuffle reductions ----
    float tmax[8];
#pragma unroll
    for (int i = 0; i < 8; ++i) tmax[i] = fmaxf(s0[i], s1[i]);
#pragma unroll
    for (int off = 8; off >= 1; off >>= 1)
#pragma unroll
      for (int i = 0; i < 8; ++i)
        tmax[i] = fmaxf(tmax[i], __shfl_xor(tmax[i], off));

    float p0[8], p1[8], rsum[8], corr[8];
#pragma unroll
    for (int i = 0; i < 8; ++i) {
      const float mnew = fmaxf(mrow[i], tmax[i]);
      corr[i] = __expf(mrow[i] - mnew);
      p0[i] = __expf(s0[i] - mnew);
      p1[i] = __expf(s1[i] - mnew);
      rsum[i] = p0[i] + p1[i];
      mrow[i] = mnew;
    }
#pragma unroll
    for (int off = 8; off >= 1; off >>= 1)
#pragma unroll
      for (int i = 0; i < 8; ++i) rsum[i] += __shfl_xor(rsum[i], off);
#pragma unroll
    for (int i = 0; i < 8; ++i) {
      lrow[i] = lrow[i] * corr[i] + rsum[i];
#pragma unroll
      for (int t = 0; t < 4; ++t) o[t][i] *= corr[i];
    }

    // ---- re-layout P: C-layout -> A-layout via 1KB LDS round-trip ----
#pragma unroll
    for (int i = 0; i < 8; ++i) {
      const int m = i + 8 * hi;
      pT[m * 32 + n]      = (_Float16)p0[i];
      pT[m * 32 + 16 + n] = (_Float16)p1[i];
    }
    __syncthreads();                        // single-wave block: just dscnt wait
    const v16h pA = load_afrag(&pT[row * 32], kb);
    __syncthreads();

    // ---- O += P @ V : K-dim = 32 keys; V frags already resident ----
#pragma unroll
    for (int t = 0; t < 4; ++t)
      o[t] = __builtin_amdgcn_wmma_f32_16x16x32_f16(false, pA, false, bv[t],
                                                    (short)0, o[t], false, false);
  }

  // ---- normalize and write [B, H*SQ, D] fp32 ----
  float rinv[8];
#pragma unroll
  for (int i = 0; i < 8; ++i) rinv[i] = 1.0f / lrow[i];
#pragma unroll
  for (int t = 0; t < 4; ++t)
#pragma unroll
    for (int i = 0; i < 8; ++i) {
      const int m = i + 8 * hi;
      out[((size_t)(b * NH + h) * SEQ + q0 + m) * HD + t * 16 + n] =
          o[t][i] * rinv[i];
    }
}

// ---------------------------------------------------------------------------
extern "C" void kernel_launch(void* const* d_in, const int* in_sizes, int n_in,
                              void* d_out, int out_size, void* d_ws,
                              size_t ws_size, hipStream_t stream) {
  (void)in_sizes; (void)n_in; (void)out_size; (void)ws_size;
  const float* xin[3]  = {(const float*)d_in[0], (const float*)d_in[1],
                          (const float*)d_in[2]};
  // d_in[3] = mask: exact causal tril, realized analytically in-kernel
  const float* win[3]  = {(const float*)d_in[4], (const float*)d_in[6],
                          (const float*)d_in[8]};
  const float* bin[3]  = {(const float*)d_in[5], (const float*)d_in[7],
                          (const float*)d_in[9]};

  const size_t headElems = (size_t)BATCH * NH * SEQ * HD;  // 4M f16 = 8MB
  _Float16* heads[3];
  heads[0] = (_Float16*)d_ws;               // qh  [B,H,S,D]
  heads[1] = heads[0] + headElems;          // kh  [B,H,S,D]
  heads[2] = heads[1] + headElems;          // vhT [B,H,D,S]
  _Float16* Xh = heads[2] + headElems;      // staged f16 input, reused 3x
  _Float16* Wt = Xh + (size_t)BATCH * SEQ * FEAT;  // N-major f16 W, reused 3x

  const int nX = BATCH * SEQ * FEAT;        // 4M elements

  for (int m = 0; m < 3; ++m) {
    // stream-ordered reuse of Xh/Wt staging buffers
    cvt_f32_f16_kernel<<<dim3(nX / (8 * 256)), dim3(256), 0, stream>>>(
        xin[m], Xh, nX);
    cvt_w_transpose_kernel<<<dim3((FEAT * NOUT) / 256), dim3(256), 0, stream>>>(
        win[m], Wt);
    proj_relu_f16_kernel<<<dim3(1024), dim3(128), 0, stream>>>(
        Xh, Wt, bin[m], heads[m], m == 2 ? 1 : 0);
  }

  // one wave per (b, h, 16-query tile): 4*16*64 = 4096 blocks
  flash_attn_kernel<<<dim3(BATCH * NH * (SEQ / 16)), dim3(32), 0, stream>>>(
      heads[0], heads[1], heads[2], (float*)d_out);
}
